// Attention_17257178595362
// MI455X (gfx1250) — matrix operand order repaired and checked by reading, not compiled, not run on hardware
//
#include <hip/hip_runtime.h>

#define TQS 2048
#define TKVS 2048
#define DMODEL 1024
#define NH 16
#define DHEAD 64
#define NBATCH 2

typedef __attribute__((ext_vector_type(16))) __bf16 v16bf;
typedef __attribute__((ext_vector_type(8)))  float  v8f;
typedef unsigned short u16;
typedef unsigned int   u32;

// ---------- helpers ----------

__device__ __forceinline__ u16 f2bf(float f) {
    union { float f; u32 u; } x; x.f = f;
    u32 u = x.u;
    return (u16)((u + 0x7FFFu + ((u >> 16) & 1u)) >> 16);  // RNE
}

// Build a 16-element bf16 fragment from two 16-byte (8 x bf16) chunks.
__device__ __forceinline__ v16bf ldfrag(const u16* p0, const u16* p1) {
    union { v16bf v; uint4 q[2]; } u;
    u.q[0] = *reinterpret_cast<const uint4*>(p0);
    u.q[1] = *reinterpret_cast<const uint4*>(p1);
    return u.v;
}

__device__ __forceinline__ v8f wmma_bf16(v16bf a, v16bf b, v8f c) {
    // D = A(16x32) * B(32x16) + C, fp32 accumulate
    return __builtin_amdgcn_wmma_f32_16x16x32_bf16(false, a, false, b, (short)0, c, false, false);
}

// ---------- prep kernels ----------

// Wt[n][k] = bf16(W[k][n]); W is 1024x1024 fp32, Wt is 1024x1024 bf16
__global__ void k_wtrans(const float* __restrict__ W, u16* __restrict__ Wt) {
    int tid = blockIdx.x * blockDim.x + threadIdx.x;   // 1M threads
    int n = tid >> 10, k = tid & 1023;
    Wt[(size_t)n * 1024 + k] = f2bf(W[(size_t)k * 1024 + n]);
}

__global__ void k_f2bf(const float* __restrict__ X, u16* __restrict__ Y, int nElem) {
    int i = blockIdx.x * blockDim.x + threadIdx.x;
    if (i < nElem) Y[i] = f2bf(X[i]);
}

// ---------- generic GEMM: C[M=4096, N=1024] = A[M,1024] * W[1024,N] + bias ----------
// A: bf16 row-major [m][k]; Wt: bf16 transposed weights [n][k].
// Wave tile 64x32 (4 M-tiles x 2 N-tiles): 8 WMMA per 12 b128 loads per K-step.
// mode 0: bf16 out at [b][h][t][f]   (Q/K layout), value = (acc+bias)*scale
// mode 1: bf16 out at [b][h][f][t]   (V transposed), value = (acc+bias)*scale
// mode 2: fp32 out row-major [m][n],  value = acc+bias
__global__ __launch_bounds__(256) void k_gemm(const u16* __restrict__ A,
                                              const u16* __restrict__ Wt,
                                              const float* __restrict__ bias,
                                              void* __restrict__ out,
                                              int mode, float scale) {
    const int lane = threadIdx.x & 31;
    const int wave = threadIdx.x >> 5;
    const int wm = wave >> 2, wn = wave & 3;            // 2x4 wave grid
    const int M0 = blockIdx.y * 128 + wm * 64;
    const int N0 = blockIdx.x * 128 + wn * 32;
    const int l15 = lane & 15;
    const int hi  = lane >> 4;                          // 0 or 1
    const int kloA = hi << 3;                           // A frag: 0/8
    const int kloB = hi << 3;                           // B frag: 0/8

    v8f acc[4][2] = {};
    const u16* Arow[4];
    #pragma unroll
    for (int mt = 0; mt < 4; ++mt)
        Arow[mt] = A + (size_t)(M0 + mt * 16 + l15) * 1024;
    const u16* Brow0 = Wt + (size_t)(N0 + l15) * 1024;
    const u16* Brow1 = Brow0 + (size_t)16 * 1024;

    for (int k0 = 0; k0 < 1024; k0 += 32) {
        v16bf b0 = ldfrag(Brow0 + k0 + kloB, Brow0 + k0 + kloB + 16);
        v16bf b1 = ldfrag(Brow1 + k0 + kloB, Brow1 + k0 + kloB + 16);
        #pragma unroll
        for (int mt = 0; mt < 4; ++mt) {
            v16bf a = ldfrag(Arow[mt] + k0 + kloA, Arow[mt] + k0 + kloA + 16);
            acc[mt][0] = wmma_bf16(a, b0, acc[mt][0]);
            acc[mt][1] = wmma_bf16(a, b1, acc[mt][1]);
        }
    }

    u16*   outU = (u16*)out;
    float* outF = (float*)out;
    #pragma unroll
    for (int mt = 0; mt < 4; ++mt) {
        #pragma unroll
        for (int nt = 0; nt < 2; ++nt) {
            v8f c = acc[mt][nt];
            int n  = N0 + nt * 16 + l15;
            float bn = bias[n];
            int mb = M0 + mt * 16 + (hi << 3);          // first row for this lane half
            if (mode == 0) {
                int bb = mb >> 11, h = n >> 6, f = n & 63;
                #pragma unroll
                for (int r = 0; r < 8; ++r) {
                    int t = (mb + r) & (TQS - 1);
                    outU[((size_t)(bb * NH + h) * TQS + t) * DHEAD + f] = f2bf((c[r] + bn) * scale);
                }
            } else if (mode == 1) {
                int bb = mb >> 11, t0 = mb & (TKVS - 1), h = n >> 6, f = n & 63;
                union { u16 s[8]; uint4 q; } pk;
                #pragma unroll
                for (int r = 0; r < 8; ++r) pk.s[r] = f2bf((c[r] + bn) * scale);
                *reinterpret_cast<uint4*>(&outU[((size_t)(bb * NH + h) * DHEAD + f) * TKVS + t0]) = pk.q;
            } else {
                #pragma unroll
                for (int r = 0; r < 8; ++r)
                    outF[(size_t)(mb + r) * DMODEL + n] = c[r] + bn;
            }
        }
    }
}

// ---------- flash attention ----------
// Q: bf16 [B,H,TQ,DH] (pre-scaled by 1/sqrt(DH)); K: bf16 [B,H,TKV,DH];
// Vt: bf16 [B,H,DH,TKV]; mask: u8 [B,TQ,TKV]; O: bf16 [B,TQ,H*DH]
__global__ __launch_bounds__(128) void k_attn(const u16* __restrict__ Q,
                                              const u16* __restrict__ K,
                                              const u16* __restrict__ Vt,
                                              const unsigned char* __restrict__ mask,
                                              u16* __restrict__ O) {
    __shared__ u16 Plds[4 * 16 * 32];                  // per-wave 16x32 bf16 P tile
    const int lane = threadIdx.x & 31;
    const int w    = threadIdx.x >> 5;
    const int l15  = lane & 15;
    const int hi   = lane >> 4;
    const int bh   = blockIdx.x >> 5;                  // 32 q-tiles of 64 per (b,h)
    const int qt   = blockIdx.x & 31;
    const int b    = bh >> 4, h = bh & 15;
    const int qbase = qt * 64 + w * 16;
    u16* Pw = Plds + w * 512;

    const u16* Qb = Q  + (size_t)bh * TQS  * DHEAD;
    const u16* Kb = K  + (size_t)bh * TKVS * DHEAD;
    const u16* Vb = Vt + (size_t)bh * DHEAD * TKVS;
    const unsigned char* Mb = mask + (size_t)b * TQS * TKVS;

    // Q A-fragments for this wave's 16 rows (features 0..31, 32..63)
    const int klo = hi << 3;
    const u16* qrow = Qb + (size_t)(qbase + l15) * DHEAD;
    v16bf qa0 = ldfrag(qrow + klo,      qrow + klo + 16);
    v16bf qa1 = ldfrag(qrow + 32 + klo, qrow + 32 + klo + 16);

    float mrow[8], lsum[8];
    v8f o[4] = {};
    #pragma unroll
    for (int r = 0; r < 8; ++r) { mrow[r] = -3.0e38f; lsum[r] = 0.f; }

    for (int kv0 = 0; kv0 < TKVS; kv0 += 32) {
        if (kv0 + 32 < TKVS)
            __builtin_prefetch(Kb + (size_t)(kv0 + 32 + l15) * DHEAD, 0, 1);

        // ---- scores: two 16x16 tiles (kv0..15, kv0+16..31) ----
        const int kfeat = hi << 3;
        const u16* kr0 = Kb + (size_t)(kv0 + l15) * DHEAD + kfeat;
        const u16* kr1 = Kb + (size_t)(kv0 + 16 + l15) * DHEAD + kfeat;
        v8f s0 = {}, s1 = {};
        s0 = wmma_bf16(qa0, ldfrag(kr0,      kr0 + 16), s0);
        s0 = wmma_bf16(qa1, ldfrag(kr0 + 32, kr0 + 48), s0);
        s1 = wmma_bf16(qa0, ldfrag(kr1,      kr1 + 16), s1);
        s1 = wmma_bf16(qa1, ldfrag(kr1 + 32, kr1 + 48), s1);

        // ---- mask + online softmax (rows live across each 16-lane half) ----
        #pragma unroll
        for (int r = 0; r < 8; ++r) {
            int qrw = qbase + r + (hi << 3);
            const unsigned char msk0 = Mb[(size_t)qrw * TKVS + kv0 + l15];
            const unsigned char msk1 = Mb[(size_t)qrw * TKVS + kv0 + 16 + l15];
            float v0 = msk0 ? s0[r] : -1.0e30f;
            float v1 = msk1 ? s1[r] : -1.0e30f;
            float sm = fmaxf(v0, v1);
            sm = fmaxf(sm, __shfl_xor(sm, 1));
            sm = fmaxf(sm, __shfl_xor(sm, 2));
            sm = fmaxf(sm, __shfl_xor(sm, 4));
            sm = fmaxf(sm, __shfl_xor(sm, 8));
            float nm   = fmaxf(mrow[r], sm);
            float corr = __expf(mrow[r] - nm);
            mrow[r] = nm;
            float p0 = __expf(v0 - nm);
            float p1 = __expf(v1 - nm);
            float ps = p0 + p1;
            ps += __shfl_xor(ps, 1);
            ps += __shfl_xor(ps, 2);
            ps += __shfl_xor(ps, 4);
            ps += __shfl_xor(ps, 8);
            lsum[r] = lsum[r] * corr + ps;
            o[0][r] *= corr; o[1][r] *= corr; o[2][r] *= corr; o[3][r] *= corr;
            int prow = r + (hi << 3);
            Pw[prow * 32 + l15]      = f2bf(p0);
            Pw[prow * 32 + 16 + l15] = f2bf(p1);
        }
        asm volatile("s_wait_dscnt 0" ::: "memory");

        // ---- P*V: read P back as A-fragment, Vt rows are contiguous B-fragments ----
        const u16* pr = Pw + l15 * 32 + klo;
        v16bf pa = ldfrag(pr, pr + 16);
        #pragma unroll
        for (int ft = 0; ft < 4; ++ft) {
            const u16* vr = Vb + (size_t)(ft * 16 + l15) * TKVS + kv0 + (hi << 3);
            o[ft] = wmma_bf16(pa, ldfrag(vr, vr + 16), o[ft]);
        }
    }

    // ---- epilogue: O[b][t][h*64+f] = o/lsum (0 for fully-masked rows) ----
    #pragma unroll
    for (int r = 0; r < 8; ++r) {
        float inv = (mrow[r] > -1.0e29f && lsum[r] > 0.f) ? (1.0f / lsum[r]) : 0.f;
        int qrw = qbase + r + (hi << 3);
        size_t rowoff = ((size_t)(b * TQS + qrw)) * DMODEL + h * DHEAD;
        #pragma unroll
        for (int ft = 0; ft < 4; ++ft)
            O[rowoff + ft * 16 + l15] = f2bf(o[ft][r] * inv);
    }
}

// ---------- launch ----------

extern "C" void kernel_launch(void* const* d_in, const int* in_sizes, int n_in,
                              void* d_out, int out_size, void* d_ws, size_t ws_size,
                              hipStream_t stream) {
    const float* inputs_q  = (const float*)d_in[0];
    const float* inputs_kv = (const float*)d_in[1];
    const unsigned char* mask = (const unsigned char*)d_in[2];
    const float* Wq = (const float*)d_in[3];
    const float* bq = (const float*)d_in[4];
    const float* Wk = (const float*)d_in[5];
    const float* bk = (const float*)d_in[6];
    const float* Wv = (const float*)d_in[7];
    const float* bv = (const float*)d_in[8];
    const float* Wo = (const float*)d_in[9];
    const float* bo = (const float*)d_in[10];

    unsigned char* ws = (unsigned char*)d_ws;
    const size_t MB = (size_t)1 << 20;
    u16* Wtq = (u16*)(ws + 0 * MB);    // 2 MB each, transposed bf16 weights
    u16* Wtk = (u16*)(ws + 2 * MB);
    u16* Wtv = (u16*)(ws + 4 * MB);
    u16* Wto = (u16*)(ws + 6 * MB);
    u16* Xq  = (u16*)(ws + 8 * MB);    // 8 MB bf16 inputs
    u16* Xkv = (u16*)(ws + 16 * MB);
    u16* Qb  = (u16*)(ws + 24 * MB);   // 8 MB bf16 [B,H,T,DH] (pre-scaled)
    u16* Kb  = (u16*)(ws + 32 * MB);   // 8 MB bf16 [B,H,T,DH]
    u16* Vt  = (u16*)(ws + 40 * MB);   // 8 MB bf16 [B,H,DH,T]
    u16* Ob  = (u16*)(ws + 48 * MB);   // 8 MB bf16 [B*T, D]

    k_wtrans<<<4096, 256, 0, stream>>>(Wq, Wtq);
    k_wtrans<<<4096, 256, 0, stream>>>(Wk, Wtk);
    k_wtrans<<<4096, 256, 0, stream>>>(Wv, Wtv);
    k_wtrans<<<4096, 256, 0, stream>>>(Wo, Wto);
    k_f2bf<<<16384, 256, 0, stream>>>(inputs_q,  Xq,  NBATCH * TQS * DMODEL);
    k_f2bf<<<16384, 256, 0, stream>>>(inputs_kv, Xkv, NBATCH * TKVS * DMODEL);

    dim3 g(8, 32);  // N/128, M/128
    k_gemm<<<g, 256, 0, stream>>>(Xq,  Wtq, bq, (void*)Qb, 0, 0.125f);  // fold 1/sqrt(64)
    k_gemm<<<g, 256, 0, stream>>>(Xkv, Wtk, bk, (void*)Kb, 0, 1.0f);
    k_gemm<<<g, 256, 0, stream>>>(Xkv, Wtv, bv, (void*)Vt, 1, 1.0f);

    k_attn<<<NBATCH * NH * (TQS / 64), 128, 0, stream>>>(Qb, Kb, Vt, mask, Ob);

    k_gemm<<<g, 256, 0, stream>>>(Ob, Wto, bo, d_out, 2, 1.0f);
}
